// Attention_48172353192538
// MI455X (gfx1250) — matrix operand order repaired and checked by reading, not compiled, not run on hardware
//
#include <hip/hip_runtime.h>
#include <hip/hip_bf16.h>

typedef __attribute__((ext_vector_type(16))) __bf16 v16bf;
typedef __attribute__((ext_vector_type(8)))  __bf16 v8bf;
typedef __attribute__((ext_vector_type(8)))  float  v8f;

#define BATCH 2
#define TSEQ  2048
#define CDIM  1024
#define NH    16
#define HD    64
#define M_ROWS (BATCH*TSEQ)        // 4096
#define N_QKV  (3*CDIM)            // 3072
#define ASIZE  (BATCH*TSEQ*CDIM)   // 4194304 floats (output a)
#define PHALF  (BATCH*NH*TSEQ*HD)  // 4194304 (one of k/v in present)

__device__ __forceinline__ v8f wmma_bf16(v16bf a, v16bf b, v8f c) {
  return __builtin_amdgcn_wmma_f32_16x16x32_bf16(false, a, false, b, (short)0, c,
                                                 false, false);
}

// A-frag: elems 0..7 = K base..base+7, elems 8..15 = K base+16..base+23 (base = g*8)
__device__ __forceinline__ v16bf load_a_frag(const __bf16* p) {
  v8bf lo = *(const v8bf*)(p);
  v8bf hi = *(const v8bf*)(p + 16);
  return __builtin_shufflevector(lo, hi, 0, 1, 2, 3, 4, 5, 6, 7,
                                 8, 9, 10, 11, 12, 13, 14, 15);
}
// B-frag: elems 0..7 = K base..base+7, elems 8..15 = K base+16..base+23 (base = g*8)
__device__ __forceinline__ v16bf load_b_frag(const __bf16* p) {
  v8bf lo = *(const v8bf*)(p);
  v8bf hi = *(const v8bf*)(p + 16);
  return __builtin_shufflevector(lo, hi, 0, 1, 2, 3, 4, 5, 6, 7,
                                 8, 9, 10, 11, 12, 13, 14, 15);
}

__global__ void cvt_kernel(const float* __restrict__ src, __bf16* __restrict__ dst, int n) {
  int i = blockIdx.x * blockDim.x + threadIdx.x;
  if (i < n) dst[i] = (__bf16)src[i];
}

// fp32 [K][N] -> bf16 [N][K] (K-contiguous per output column => B-frags vectorize)
__global__ void cvt_transpose_kernel(const float* __restrict__ src, __bf16* __restrict__ dst,
                                     int K, int N) {
  int i = blockIdx.x * blockDim.x + threadIdx.x;
  if (i < K * N) {
    int k = i / N, n = i - k * N;
    dst[(size_t)n * K + k] = (__bf16)src[i];
  }
}

// qkv = x @ c_attn_w + b ; scatter q/k (bf16 [B,H,T,hd]), v (bf16 [B,H,hd,T]), fp32 present
__global__ void qkv_gemm_kernel(const __bf16* __restrict__ xb, const __bf16* __restrict__ wt,
                                const float* __restrict__ bias,
                                __bf16* __restrict__ qb, __bf16* __restrict__ kb,
                                __bf16* __restrict__ vtb, float* __restrict__ present) {
  const int lane = threadIdx.x & 31;
  const int wave = threadIdx.x >> 5;
  const int col0 = (blockIdx.x * 4 + wave) * 64;   // 16x64 strip per wave
  const int row0 = blockIdx.y * 16;
  const int r = lane & 15, g = lane >> 4;
  v8f acc[4] = {};
  const __bf16* arow = xb + (size_t)(row0 + r) * CDIM + g * 8;
  const __bf16* b0 = wt + (size_t)(col0 + r) * CDIM + g * 8;
  for (int k0 = 0; k0 < CDIM; k0 += 32) {
    __builtin_prefetch(b0 + k0 + 512, 0, 3);
    v16bf af = load_a_frag(arow + k0);
    acc[0] = wmma_bf16(af, load_b_frag(b0 + k0), acc[0]);
    acc[1] = wmma_bf16(af, load_b_frag(b0 + (size_t)16 * CDIM + k0), acc[1]);
    acc[2] = wmma_bf16(af, load_b_frag(b0 + (size_t)32 * CDIM + k0), acc[2]);
    acc[3] = wmma_bf16(af, load_b_frag(b0 + (size_t)48 * CDIM + k0), acc[3]);
  }
#pragma unroll
  for (int jt = 0; jt < 4; ++jt) {
    const int col = col0 + jt * 16 + r;
    const int j  = col >> 10;                 // 0=q 1=k 2=v
    const int cc = col & (CDIM - 1);
    const int h  = cc >> 6, d = cc & (HD - 1);
    const float bv = bias[col];
#pragma unroll
    for (int rr = 0; rr < 8; ++rr) {
      const int m = row0 + g * 8 + rr;
      const int b = m >> 11;
      const int t = m & (TSEQ - 1);
      const float v = acc[jt][rr] + bv;
      const int bh = b * NH + h;
      if (j == 0) {
        qb[((size_t)bh * TSEQ + t) * HD + d] = (__bf16)v;
      } else if (j == 1) {
        kb[((size_t)bh * TSEQ + t) * HD + d] = (__bf16)v;
        present[((size_t)bh * TSEQ + t) * HD + d] = v;
      } else {
        vtb[((size_t)bh * HD + d) * TSEQ + t] = (__bf16)v;   // head-transposed V
        present[PHALF + ((size_t)bh * TSEQ + t) * HD + d] = v;
      }
    }
  }
}

// flash attention: one wave = one (b,h, 16-row q tile)
__global__ void attn_kernel(const __bf16* __restrict__ qb, const __bf16* __restrict__ kb,
                            const __bf16* __restrict__ vtb, __bf16* __restrict__ ab) {
  __shared__ __attribute__((aligned(16))) __bf16 pbuf[4][16 * 32];
  const int lane = threadIdx.x & 31;
  const int wave = threadIdx.x >> 5;
  __bf16* pw = pbuf[wave];
  const int task = blockIdx.x * 4 + wave;       // B*NH*(T/16) = 4096 tasks
  const int qt = task & 127;
  const int bh = task >> 7;
  const int bidx = bh >> 4;
  const int h = bh & 15;
  const __bf16* Q  = qb  + (size_t)bh * TSEQ * HD;
  const __bf16* K  = kb  + (size_t)bh * TSEQ * HD;
  const __bf16* Vt = vtb + (size_t)bh * HD * TSEQ;   // [d][t]
  const int q0 = qt * 16;
  const int r = lane & 15, g = lane >> 4;

  const v16bf qf0 = load_a_frag(Q + (q0 + r) * HD + g * 8);        // d 0..31
  const v16bf qf1 = load_a_frag(Q + (q0 + r) * HD + 32 + g * 8);   // d 32..63

  v8f acc[4] = {};
  float mrow[8], lrow[8];
#pragma unroll
  for (int rr = 0; rr < 8; ++rr) { mrow[rr] = -3.0e38f; lrow[rr] = 0.f; }

  const int nchunks = (q0 + 16 + 31) >> 5;       // 32-key chunks, causal bound
  for (int c = 0; c < nchunks; ++c) {
    const int k0 = c * 32;
    float s0[8], s1[8];
#pragma unroll
    for (int half = 0; half < 2; ++half) {
      const int kb0 = k0 + half * 16;
      const v16bf kf0 = load_b_frag(K + (kb0 + r) * HD + g * 8);        // d 0..31
      const v16bf kf1 = load_b_frag(K + (kb0 + r) * HD + 32 + g * 8);   // d 32..63
      v8f s = {};
      s = wmma_bf16(qf0, kf0, s);
      s = wmma_bf16(qf1, kf1, s);
#pragma unroll
      for (int rr = 0; rr < 8; ++rr) {
        const int qrow = q0 + g * 8 + rr;
        const int kcol = kb0 + r;
        float sv = s[rr] * 0.125f;               // 1/sqrt(64)
        if (kcol > qrow) sv = -1.0e10f;          // causal mask (matches reference)
        if (half == 0) s0[rr] = sv; else s1[rr] = sv;
      }
    }
    // row max over this 32-key chunk (reduce across 16-lane half-wave)
    float cm[8];
#pragma unroll
    for (int rr = 0; rr < 8; ++rr) cm[rr] = fmaxf(s0[rr], s1[rr]);
#pragma unroll
    for (int off = 1; off <= 8; off <<= 1)
#pragma unroll
      for (int rr = 0; rr < 8; ++rr) cm[rr] = fmaxf(cm[rr], __shfl_xor(cm[rr], off, 32));

    float psum[8];
#pragma unroll
    for (int rr = 0; rr < 8; ++rr) {
      const float mnew = fmaxf(mrow[rr], cm[rr]);
      const float sc = __expf(mrow[rr] - mnew);
      const float p0 = __expf(s0[rr] - mnew);
      const float p1 = __expf(s1[rr] - mnew);
      mrow[rr] = mnew;
      acc[0][rr] *= sc; acc[1][rr] *= sc; acc[2][rr] *= sc; acc[3][rr] *= sc;
      lrow[rr] *= sc;
      psum[rr] = p0 + p1;
      pw[(g * 8 + rr) * 32 + r]      = (__bf16)p0;   // stage P for A-frag reload
      pw[(g * 8 + rr) * 32 + 16 + r] = (__bf16)p1;
    }
#pragma unroll
    for (int off = 1; off <= 8; off <<= 1)
#pragma unroll
      for (int rr = 0; rr < 8; ++rr) psum[rr] += __shfl_xor(psum[rr], off, 32);
#pragma unroll
    for (int rr = 0; rr < 8; ++rr) lrow[rr] += psum[rr];

    // P(16x32) @ V(32x64): A-frag via LDS (DS in-order per wave), B-frags from Vt rows
    const v16bf pf = load_a_frag(pw + r * 32 + g * 8);
#pragma unroll
    for (int jt = 0; jt < 4; ++jt) {
      const v16bf vf = load_b_frag(Vt + (size_t)(jt * 16 + r) * TSEQ + k0 + g * 8);
      acc[jt] = wmma_bf16(pf, vf, acc[jt]);
    }
  }
  // merged-head output [B,T,C] in bf16 for the proj GEMM
#pragma unroll
  for (int rr = 0; rr < 8; ++rr) {
    const int t = q0 + g * 8 + rr;
    const float inv = 1.f / lrow[rr];
    __bf16* dst = ab + ((size_t)(bidx * TSEQ + t)) * CDIM + h * HD;
#pragma unroll
    for (int jt = 0; jt < 4; ++jt) dst[jt * 16 + r] = (__bf16)(acc[jt][rr] * inv);
  }
}

// out = attn_merged @ c_proj_w + b  (fp32 output), 16x64 strip per wave
__global__ void proj_gemm_kernel(const __bf16* __restrict__ ab, const __bf16* __restrict__ wt,
                                 const float* __restrict__ bias, float* __restrict__ out) {
  const int lane = threadIdx.x & 31;
  const int wave = threadIdx.x >> 5;
  const int col0 = (blockIdx.x * 4 + wave) * 64;
  const int row0 = blockIdx.y * 16;
  const int r = lane & 15, g = lane >> 4;
  v8f acc[4] = {};
  const __bf16* arow = ab + (size_t)(row0 + r) * CDIM + g * 8;
  const __bf16* b0 = wt + (size_t)(col0 + r) * CDIM + g * 8;
  for (int k0 = 0; k0 < CDIM; k0 += 32) {
    __builtin_prefetch(b0 + k0 + 512, 0, 3);
    v16bf af = load_a_frag(arow + k0);
    acc[0] = wmma_bf16(af, load_b_frag(b0 + k0), acc[0]);
    acc[1] = wmma_bf16(af, load_b_frag(b0 + (size_t)16 * CDIM + k0), acc[1]);
    acc[2] = wmma_bf16(af, load_b_frag(b0 + (size_t)32 * CDIM + k0), acc[2]);
    acc[3] = wmma_bf16(af, load_b_frag(b0 + (size_t)48 * CDIM + k0), acc[3]);
  }
#pragma unroll
  for (int jt = 0; jt < 4; ++jt) {
    const float bv = bias[col0 + jt * 16 + r];
#pragma unroll
    for (int rr = 0; rr < 8; ++rr) {
      const int m = row0 + g * 8 + rr;
      out[(size_t)m * CDIM + col0 + jt * 16 + r] = acc[jt][rr] + bv;
    }
  }
}

extern "C" void kernel_launch(void* const* d_in, const int* in_sizes, int n_in,
                              void* d_out, int out_size, void* d_ws, size_t ws_size,
                              hipStream_t stream) {
  const float* x      = (const float*)d_in[0];
  const float* attn_w = (const float*)d_in[1];
  const float* attn_b = (const float*)d_in[2];
  const float* proj_w = (const float*)d_in[3];
  const float* proj_b = (const float*)d_in[4];
  float* out = (float*)d_out;

  // workspace carve-up (bf16): ~50 MB
  __bf16* xb    = (__bf16*)d_ws;
  __bf16* wqkvT = xb    + (size_t)M_ROWS * CDIM;       // [3072][1024]
  __bf16* wprojT = wqkvT + (size_t)CDIM * N_QKV;       // [1024][1024]
  __bf16* qbuf  = wprojT + (size_t)CDIM * CDIM;        // [B,H,T,hd]
  __bf16* kbuf  = qbuf  + (size_t)PHALF;               // [B,H,T,hd]
  __bf16* vtbuf = kbuf  + (size_t)PHALF;               // [B,H,hd,T]
  __bf16* abuf  = vtbuf + (size_t)PHALF;               // [B,T,C]

  cvt_kernel<<<(M_ROWS * CDIM + 255) / 256, 256, 0, stream>>>(x, xb, M_ROWS * CDIM);
  cvt_transpose_kernel<<<(CDIM * N_QKV + 255) / 256, 256, 0, stream>>>(attn_w, wqkvT,
                                                                       CDIM, N_QKV);
  cvt_transpose_kernel<<<(CDIM * CDIM + 255) / 256, 256, 0, stream>>>(proj_w, wprojT,
                                                                      CDIM, CDIM);

  qkv_gemm_kernel<<<dim3(N_QKV / 256, M_ROWS / 16), 128, 0, stream>>>(
      xb, wqkvT, attn_b, qbuf, kbuf, vtbuf, out + ASIZE);

  attn_kernel<<<(BATCH * NH * (TSEQ / 16)) / 4, 128, 0, stream>>>(qbuf, kbuf, vtbuf, abuf);

  proj_gemm_kernel<<<dim3(CDIM / 256, M_ROWS / 16), 128, 0, stream>>>(abuf, wprojT,
                                                                      proj_b, out);
}
